// CharRNN_90658169684120
// MI455X (gfx1250) — matrix operand order repaired and checked by reading, not compiled, not run on hardware
//
#include <hip/hip_runtime.h>
#include <hip/hip_bf16.h>
#include <stdint.h>

// ---------------- problem constants ----------------
#define Bq 32
#define Tq 256
#define Hq 1024
#define Vq 256
#define G4 4096            // 4*Hq
#define KT 32              // K tiles per H (Hq/32)
#define NWG 24             // persistent workgroups for the scan
#define THREADS 256

typedef __attribute__((ext_vector_type(16))) __bf16           v16bf;
typedef __attribute__((ext_vector_type(8)))  float            v8f;
typedef __attribute__((ext_vector_type(8)))  unsigned short   v8u;
typedef __attribute__((ext_vector_type(16))) unsigned short   v16u;

union BFV { v16u u; v16bf b; };

__device__ __forceinline__ unsigned short f2bf(float f) {
  unsigned int u = __float_as_uint(f);
  u += 0x7fffu + ((u >> 16) & 1u);           // round-to-nearest-even
  return (unsigned short)(u >> 16);
}

__device__ __forceinline__ float sigm(float x) { return 1.0f / (1.0f + __expf(-x)); }

__device__ __forceinline__ v8f vzero() {
  v8f z = {0.f, 0.f, 0.f, 0.f, 0.f, 0.f, 0.f, 0.f};
  return z;
}

// Build a v16bf fragment from two contiguous 16-byte chunks (global or LDS).
__device__ __forceinline__ v16bf frag16(const unsigned short* lo, const unsigned short* hi) {
  v8u a = *reinterpret_cast<const v8u*>(lo);
  v8u b = *reinterpret_cast<const v8u*>(hi);
  BFV f;
#pragma unroll
  for (int i = 0; i < 8; ++i) { f.u[i] = a[i]; f.u[i + 8] = b[i]; }
  return f.b;
}

__device__ __forceinline__ v8f wmma_bf16(v16bf a, v16bf b, v8f c) {
  return __builtin_amdgcn_wmma_f32_16x16x32_bf16(false, a, false, b, (short)0, c, false, false);
}

// A-fragment (16x32 bf16) from row-major [rows][Hq] activation matrix.
// lanes 0-15: row M=mbase+lane, K {0..7,16..23}; lanes 16-31: same row, K {8..15,24..31}
__device__ __forceinline__ v16bf afrag(const unsigned short* m, int mbase, int kt, int lane) {
  int row = mbase + (lane & 15);
  int k0  = kt * 32 + ((lane >> 4) << 3);
  const unsigned short* p = m + (size_t)row * Hq + k0;
  return frag16(p, p + 16);
}

// B-fragment (32x16 bf16) from pre-packed weights: lane l -> col nt*16+(l&15),
// K {0..7,16..23} (l<16) or {8..15,24..31} (l>=16) within tile kt. 32 bytes/lane contiguous.
__device__ __forceinline__ v16bf bfrag(const unsigned short* wpk, int nt, int kt, int ktt, int lane) {
  const unsigned short* p = wpk + (((size_t)nt * ktt + kt) * 32 + lane) * 16;
  return frag16(p, p + 8);
}

// Cooperative stage of a [Bq x Hq] bf16 activation matrix into LDS (64 KB).
__device__ __forceinline__ void stage_h(const unsigned short* __restrict__ src,
                                        unsigned short* dst) {
  const int n16 = (Bq * Hq) / 8;   // 4096 chunks of 16 bytes
  for (int i = threadIdx.x; i < n16; i += THREADS) {
    *reinterpret_cast<v8u*>(dst + (size_t)i * 8) =
        *reinterpret_cast<const v8u*>(src + (size_t)i * 8);
  }
}

// ---------------- grid-wide barrier (sense-reversing, agent scope) ----------------
__device__ __forceinline__ void grid_barrier(unsigned* bar, unsigned nwg) {
  __threadfence();
  __syncthreads();
  if (threadIdx.x == 0) {
    unsigned* cnt = bar;
    unsigned* gen = bar + 1;
    unsigned g = __hip_atomic_load(gen, __ATOMIC_ACQUIRE, __HIP_MEMORY_SCOPE_AGENT);
    unsigned arrived =
        __hip_atomic_fetch_add(cnt, 1u, __ATOMIC_ACQ_REL, __HIP_MEMORY_SCOPE_AGENT) + 1u;
    if (arrived == nwg) {
      __hip_atomic_store(cnt, 0u, __ATOMIC_RELAXED, __HIP_MEMORY_SCOPE_AGENT);
      __hip_atomic_fetch_add(gen, 1u, __ATOMIC_ACQ_REL, __HIP_MEMORY_SCOPE_AGENT);
    } else {
      while (__hip_atomic_load(gen, __ATOMIC_ACQUIRE, __HIP_MEMORY_SCOPE_AGENT) == g) {
        __builtin_amdgcn_s_sleep(2);
      }
    }
  }
  __syncthreads();
}

// ---------------- weight packing: f32 [K x N] row-major -> bf16 B-fragment order ----------------
__global__ void pack_b_kernel(const float* __restrict__ W, unsigned short* __restrict__ dst,
                              int K, int N) {
  size_t total = (size_t)K * N;
  int ktt = K / 32;
  for (size_t p = (size_t)blockIdx.x * blockDim.x + threadIdx.x; p < total;
       p += (size_t)gridDim.x * blockDim.x) {
    int e    = (int)(p & 15);
    int lane = (int)((p >> 4) & 31);
    size_t tile = p >> 9;                 // 512 elements per 32x16 tile
    int kt = (int)(tile % ktt);
    int nt = (int)(tile / ktt);
    int n = nt * 16 + (lane & 15);
    int k = kt * 32 + ((lane >> 4) << 3) + (e < 8 ? e : e + 8);
    dst[p] = f2bf(W[(size_t)k * N + n]);
  }
}

// ---------------- prep: emb->bf16, scan state init, barrier init ----------------
__global__ void prep_kernel(const float* __restrict__ emb,
                            const float* __restrict__ h0, const float* __restrict__ c0,
                            const float* __restrict__ h1, const float* __restrict__ c1,
                            unsigned short* __restrict__ embbf,
                            unsigned short* __restrict__ h0bf, unsigned short* __restrict__ h1bf,
                            float* __restrict__ c0buf, float* __restrict__ c1buf,
                            float* __restrict__ h0f, float* __restrict__ h1f,
                            unsigned* __restrict__ bar) {
  size_t i = (size_t)blockIdx.x * blockDim.x + threadIdx.x;
  size_t stride = (size_t)gridDim.x * blockDim.x;
  for (size_t p = i; p < (size_t)Vq * Hq; p += stride) embbf[p] = f2bf(emb[p]);
  for (size_t p = i; p < (size_t)Bq * Hq; p += stride) {
    h0bf[p] = f2bf(h0[p]);      // buffer 0 (read at t=0)
    h1bf[p] = f2bf(h1[p]);
    c0buf[p] = c0[p];
    c1buf[p] = c1[p];
    h0f[p] = h0[p];
    h1f[p] = h1[p];
  }
  if (i == 0) { bar[0] = 0u; bar[1] = 0u; }
}

// ---------------- X0 = embed(targets) @ W_ih0 + b0   (fully parallel) ----------------
// M = B*T = 8192 rows, m = t*32 + b  (X0 stored [T][B][4096] f32).
// Each wave: one column tile nt, 8 M-tiles (B-fragment reused 8x).
__global__ __launch_bounds__(THREADS)
void x0_gemm_kernel(const long long* __restrict__ targets,
                    const unsigned short* __restrict__ embbf,
                    const unsigned short* __restrict__ wih0pk,
                    const float* __restrict__ b0,
                    float* __restrict__ X0) {
  const int lane = threadIdx.x & 31;
  const int gw = (int)((blockIdx.x * blockDim.x + threadIdx.x) >> 5);
  const int nwaves = (int)((gridDim.x * blockDim.x) >> 5);
  const int NT = G4 / 16;                   // 256 column tiles
  const int MG = (Bq * Tq) / (16 * 8);      // 64 groups of 8 m-tiles

  for (int unit = gw; unit < NT * MG; unit += nwaves) {
    int nt = unit % NT;
    int mg = unit / NT;

    const unsigned short* rp[8];
#pragma unroll
    for (int j = 0; j < 8; ++j) {
      int mt = mg * 8 + j;
      int r  = mt * 16 + (lane & 15);
      int t  = r >> 5, b = r & 31;
      long long tgt = targets[(size_t)b * Tq + t];
      rp[j] = embbf + (size_t)tgt * Hq;
    }

    v8f acc[8];
#pragma unroll
    for (int j = 0; j < 8; ++j) acc[j] = vzero();

    const int khalf = (lane >> 4) << 3;
#pragma unroll 2
    for (int kt = 0; kt < KT; ++kt) {
      v16bf bf = bfrag(wih0pk, nt, kt, KT, lane);
      int k0 = kt * 32 + khalf;
#pragma unroll
      for (int j = 0; j < 8; ++j) {
        const unsigned short* p = rp[j] + k0;
        v16bf af = frag16(p, p + 16);
        acc[j] = wmma_bf16(af, bf, acc[j]);
      }
    }

    int col = nt * 16 + (lane & 15);
    float bias = b0[col];
    int rhalf = (lane >> 4) << 3;
#pragma unroll
    for (int j = 0; j < 8; ++j) {
      int mt = mg * 8 + j;
#pragma unroll
      for (int i = 0; i < 8; ++i) {
        int mrow = mt * 16 + i + rhalf;
        X0[(size_t)mrow * G4 + col] = acc[j][i] + bias;
      }
    }
  }
}

// ---------------- logits tile: h1(s) @ W_out + b_out -> out[B,T,V] ----------------
__device__ __forceinline__ void emit_logits(int s, int id, int lane,
                                            const unsigned short* h1bf,
                                            const unsigned short* woutpk,
                                            const float* __restrict__ bout,
                                            float* __restrict__ out) {
  int nt = id & 15, half = id >> 4;
  const unsigned short* h = h1bf + (size_t)((s & 1) ^ 1) * (Bq * Hq);
  v8f acc = vzero();
#pragma unroll 4
  for (int kt = 0; kt < KT; ++kt) {
    v16bf a  = afrag(h, half * 16, kt, lane);
    v16bf bf = bfrag(woutpk, nt, kt, KT, lane);
    acc = wmma_bf16(a, bf, acc);
  }
  int col = nt * 16 + (lane & 15);
  float bias = bout[col];
  int rhalf = (lane >> 4) << 3;
#pragma unroll
  for (int i = 0; i < 8; ++i) {
    int row = half * 16 + i + rhalf;                      // row == batch b
    out[((size_t)row * Tq + s) * Vq + col] = acc[i] + bias;
  }
}

// ---------------- persistent sequential scan ----------------
// 24 WGs x 8 waves. Blocks 0..15 (waves 0..127): (jt, m-half) cell tiles with activations
// staged in 128 KB of LDS; blocks 16..19 (waves 128..159): logits (1 step behind).
__global__ __launch_bounds__(THREADS)
void scan_kernel(const float* __restrict__ X0,
                 const unsigned short* __restrict__ whh0pk,
                 const unsigned short* __restrict__ wih1pk,
                 const unsigned short* __restrict__ whh1pk,
                 const unsigned short* __restrict__ woutpk,
                 const float* __restrict__ b1, const float* __restrict__ bout,
                 unsigned short* __restrict__ h0bf, unsigned short* __restrict__ h1bf,
                 float* __restrict__ c0, float* __restrict__ c1,
                 float* __restrict__ h0f, float* __restrict__ h1f,
                 float* __restrict__ out, unsigned* __restrict__ bar) {
  extern __shared__ unsigned short smem[];            // 2 x 32x1024 bf16 = 128 KB
  unsigned short* sh0 = smem;                         // layer-0 activation (h0 prev/cur)
  unsigned short* sh1 = smem + (size_t)Bq * Hq;       // layer-1 previous hidden

  const int lane = threadIdx.x & 31;
  const int gw = blockIdx.x * 8 + (threadIdx.x >> 5);
  const bool cellwg = (blockIdx.x < 16);              // uniform per block
  const int HB = Bq * Hq;

  for (int t = 0; t < Tq; ++t) {
    const int r = t & 1, w = r ^ 1;       // h(t-1) in buf[r], h(t) written to buf[w]

    // ---------- Phase A: layer-0 cell; logit waves emit step t-1 ----------
    if (cellwg) {
      // sh0 already holds h0(t-1): staged by phase B of step t-1. Only t=0 needs a stage.
      if (t == 0) {
        stage_h(h0bf + (size_t)r * HB, sh0);
        __syncthreads();
      }
      const int jt = gw >> 1, half = gw & 1;
      v8f acc[4];
#pragma unroll
      for (int g = 0; g < 4; ++g) acc[g] = vzero();
#pragma unroll 4
      for (int kt = 0; kt < KT; ++kt) {
        v16bf a = afrag(sh0, half * 16, kt, lane);    // LDS
#pragma unroll
        for (int g = 0; g < 4; ++g) {
          v16bf bf = bfrag(whh0pk, g * 64 + jt, kt, KT, lane);
          acc[g] = wmma_bf16(a, bf, acc[g]);
        }
      }
      int col = jt * 16 + (lane & 15);
      int rhalf = (lane >> 4) << 3;
      const float* x0t = X0 + (size_t)t * Bq * G4;
      unsigned short* hw = h0bf + (size_t)w * HB;
#pragma unroll
      for (int i = 0; i < 8; ++i) {
        int row = half * 16 + i + rhalf;
        const float* xr = x0t + (size_t)row * G4;
        float zi = acc[0][i] + xr[col];
        float zf = acc[1][i] + xr[1024 + col];
        float zg = acc[2][i] + xr[2048 + col];
        float zo = acc[3][i] + xr[3072 + col];
        size_t idx = (size_t)row * Hq + col;
        float cc = c0[idx];
        float cn = sigm(zf) * cc + sigm(zi) * tanhf(zg);
        float hn = sigm(zo) * tanhf(cn);
        c0[idx] = cn;
        hw[idx] = f2bf(hn);
        if (t == Tq - 1) h0f[idx] = hn;
      }
    } else if (gw >= 128 && gw < 160 && t > 0) {
      emit_logits(t - 1, gw - 128, lane, h1bf, woutpk, bout, out);
    }
    grid_barrier(bar, NWG);

    // ---------- Phase B: layer-1 cell (z1 = h0'@W_ih1 + h1@W_hh1 + b1) ----------
    if (cellwg) {
      // stage h0(t) (also serves as next step's phase-A input) and h1(t-1)
      stage_h(h0bf + (size_t)w * HB, sh0);
      stage_h(h1bf + (size_t)r * HB, sh1);
      __syncthreads();

      const int jt = gw >> 1, half = gw & 1;
      v8f acc[4];
#pragma unroll
      for (int g = 0; g < 4; ++g) acc[g] = vzero();
#pragma unroll 2
      for (int kt = 0; kt < KT; ++kt) {
        v16bf a0 = afrag(sh0, half * 16, kt, lane);   // LDS
        v16bf a1 = afrag(sh1, half * 16, kt, lane);   // LDS
#pragma unroll
        for (int g = 0; g < 4; ++g) {
          v16bf bi = bfrag(wih1pk, g * 64 + jt, kt, KT, lane);
          acc[g] = wmma_bf16(a0, bi, acc[g]);
          v16bf bh = bfrag(whh1pk, g * 64 + jt, kt, KT, lane);
          acc[g] = wmma_bf16(a1, bh, acc[g]);
        }
      }
      int col = jt * 16 + (lane & 15);
      int rhalf = (lane >> 4) << 3;
      unsigned short* hw = h1bf + (size_t)w * HB;
#pragma unroll
      for (int i = 0; i < 8; ++i) {
        int row = half * 16 + i + rhalf;
        float zi = acc[0][i] + b1[col];
        float zf = acc[1][i] + b1[1024 + col];
        float zg = acc[2][i] + b1[2048 + col];
        float zo = acc[3][i] + b1[3072 + col];
        size_t idx = (size_t)row * Hq + col;
        float cc = c1[idx];
        float cn = sigm(zf) * cc + sigm(zi) * tanhf(zg);
        float hn = sigm(zo) * tanhf(cn);
        c1[idx] = cn;
        hw[idx] = f2bf(hn);
        if (t == Tq - 1) h1f[idx] = hn;
      }
    }
    grid_barrier(bar, NWG);
  }

  // final logits for t = Tq-1
  if (gw >= 128 && gw < 160) {
    emit_logits(Tq - 1, gw - 128, lane, h1bf, woutpk, bout, out);
  }

  // final state copy: d_out tail = [h0f, c0f, h1f, c1f]
  const int tid = (int)(blockIdx.x * blockDim.x + threadIdx.x);
  const int nthreads = (int)(gridDim.x * blockDim.x);
  float* tail = out + (size_t)Bq * Tq * Vq;
  for (int i = tid; i < HB; i += nthreads) {
    tail[i]          = h0f[i];
    tail[HB + i]     = c0[i];
    tail[2 * HB + i] = h1f[i];
    tail[3 * HB + i] = c1[i];
  }
}

// ---------------- host launcher ----------------
extern "C" void kernel_launch(void* const* d_in, const int* in_sizes, int n_in,
                              void* d_out, int out_size, void* d_ws, size_t ws_size,
                              hipStream_t stream) {
  (void)in_sizes; (void)n_in; (void)out_size; (void)ws_size;
  // inputs: 0 x, 1 targets, 2 h0, 3 c0, 4 h1, 5 c1, 6 emb, 7 W_ih0, 8 W_hh0,
  //         9 b0, 10 W_ih1, 11 W_hh1, 12 b1, 13 W_out, 14 b_out
  const long long* targets = (const long long*)d_in[1];
  const float* h0_in = (const float*)d_in[2];
  const float* c0_in = (const float*)d_in[3];
  const float* h1_in = (const float*)d_in[4];
  const float* c1_in = (const float*)d_in[5];
  const float* emb   = (const float*)d_in[6];
  const float* W_ih0 = (const float*)d_in[7];
  const float* W_hh0 = (const float*)d_in[8];
  const float* b0    = (const float*)d_in[9];
  const float* W_ih1 = (const float*)d_in[10];
  const float* W_hh1 = (const float*)d_in[11];
  const float* b1    = (const float*)d_in[12];
  const float* W_out = (const float*)d_in[13];
  const float* b_out = (const float*)d_in[14];
  float* out = (float*)d_out;

  char* ws = (char*)d_ws;
  size_t off = 0;
  auto take = [&](size_t bytes) -> void* {
    void* p = ws + off;
    off += (bytes + 255) & ~(size_t)255;
    return p;
  };
  float*          X0      = (float*)take((size_t)Tq * Bq * G4 * 4);       // 134 MB
  unsigned short* wih0pk  = (unsigned short*)take((size_t)Hq * G4 * 2);   // 8 MB
  unsigned short* whh0pk  = (unsigned short*)take((size_t)Hq * G4 * 2);
  unsigned short* wih1pk  = (unsigned short*)take((size_t)Hq * G4 * 2);
  unsigned short* whh1pk  = (unsigned short*)take((size_t)Hq * G4 * 2);
  unsigned short* woutpk  = (unsigned short*)take((size_t)Hq * Vq * 2);
  unsigned short* embbf   = (unsigned short*)take((size_t)Vq * Hq * 2);
  unsigned short* h0bf    = (unsigned short*)take((size_t)2 * Bq * Hq * 2);
  unsigned short* h1bf    = (unsigned short*)take((size_t)2 * Bq * Hq * 2);
  float*          c0buf   = (float*)take((size_t)Bq * Hq * 4);
  float*          c1buf   = (float*)take((size_t)Bq * Hq * 4);
  float*          h0f     = (float*)take((size_t)Bq * Hq * 4);
  float*          h1f     = (float*)take((size_t)Bq * Hq * 4);
  unsigned*       bar     = (unsigned*)take(256);

  // 1) pack weights into WMMA B-fragment order (bf16)
  pack_b_kernel<<<2048, THREADS, 0, stream>>>(W_ih0, wih0pk, Hq, G4);
  pack_b_kernel<<<2048, THREADS, 0, stream>>>(W_hh0, whh0pk, Hq, G4);
  pack_b_kernel<<<2048, THREADS, 0, stream>>>(W_ih1, wih1pk, Hq, G4);
  pack_b_kernel<<<2048, THREADS, 0, stream>>>(W_hh1, whh1pk, Hq, G4);
  pack_b_kernel<<<128,  THREADS, 0, stream>>>(W_out, woutpk, Hq, Vq);

  // 2) embedding -> bf16, scan state + barrier init
  prep_kernel<<<512, THREADS, 0, stream>>>(emb, h0_in, c0_in, h1_in, c1_in,
                                           embbf, h0bf, h1bf, c0buf, c1buf, h0f, h1f, bar);

  // 3) hoisted input GEMM: X0[t][b][:] = emb[targets[b,t]] @ W_ih0 + b0
  x0_gemm_kernel<<<512, THREADS, 0, stream>>>(targets, embbf, wih0pk, b0, X0);

  // 4) persistent recurrent scan (grid barriers), LDS-staged activations,
  //    logits one step behind layer-0
  scan_kernel<<<NWG, THREADS, 2 * Bq * Hq * sizeof(unsigned short), stream>>>(
      X0, whh0pk, wih1pk, whh1pk, woutpk, b1, b_out, h0bf, h1bf, c0buf, c1buf,
      h0f, h1f, out, bar);
}
